// ScorePredictor_direct_88957362635113
// MI455X (gfx1250) — matrix operand order repaired and checked by reading, not compiled, not run on hardware
//
#include <hip/hip_runtime.h>
#include <hip/hip_bf16.h>

typedef __attribute__((ext_vector_type(16))) __bf16        v16bf;
typedef __attribute__((ext_vector_type(8)))  float         v8f;
typedef __attribute__((ext_vector_type(4)))  unsigned int  v4u;

#define KP      224   // concat dim padded to multiple of 32
#define CDIM    214
#define HID     128
#define H2      32
#define EPB     128   // edges per block
#define THREADS 256

// LDS partition (ushort units)
#define OFF_A   0
#define OFF_W1  (EPB * KP)              // 28672
#define OFF_H1  (OFF_W1 + HID * KP)     // 57344
#define OFF_W2  (OFF_H1 + EPB * HID)    // 73728
#define OFF_H2  (OFF_W2 + H2 * HID)     // 77824
#define SMEM_USHORTS (OFF_H2 + EPB * H2)   // 81920
#define SMEM_BYTES   (SMEM_USHORTS * 2)    // 163840

// Native f32 -> bf16 (RNE) via hardware convert; packed pair into one dword.
__device__ __forceinline__ unsigned short f2bf(float x) {
  return __builtin_bit_cast(unsigned short, (__bf16)x);
}
__device__ __forceinline__ unsigned int pack2bf(float lo, float hi) {
  const unsigned int l = f2bf(lo);
  const unsigned int h = f2bf(hi);
  return l | (h << 16);
}

union Frag { v16bf v; v4u q[2]; };

// A fragment (16x32 bf16, MxK), ISA 7.12.2 layout:
// lanes 0-15: M=lane; elems 0..7 = K 0..7, elems 8..15 = K 16..23
// lanes 16-31: M=lane-16; same pattern shifted by +8 in K
// Each half is 16 contiguous bytes in LDS -> two ds_load_b128.
__device__ __forceinline__ v16bf load_a_frag(const unsigned short* base,
                                             int mbase, int kbase, int kstride) {
  const int lane = threadIdx.x & 31;
  const int m = mbase + (lane & 15);
  const int khalf = (lane < 16) ? 0 : 8;
  const unsigned short* p = base + m * kstride + kbase + khalf;
  Frag f;
  f.q[0] = *(const v4u*)(p);        // K khalf+0..7
  f.q[1] = *(const v4u*)(p + 16);   // K khalf+16..23
  return f.v;
}

// B fragment (32x16 bf16, KxN) from weights stored [N][K] in LDS:
// lanes 0-15: N=lane, elems 0..7 = K 0..7, elems 8..15 = K 16..23;
// lanes 16-31: N=lane-16, same pattern shifted by +8 in K -> two ds_load_b128.
__device__ __forceinline__ v16bf load_b_frag(const unsigned short* base,
                                             int nbase, int kbase, int kstride) {
  const int lane = threadIdx.x & 31;
  const int n = nbase + (lane & 15);
  const int khalf = (lane < 16) ? 0 : 8;
  const unsigned short* p = base + n * kstride + kbase + khalf;
  Frag f;
  f.q[0] = *(const v4u*)(p);
  f.q[1] = *(const v4u*)(p + 16);
  return f.v;
}

__device__ __forceinline__ v8f wmma_bf16(v16bf a, v16bf b, v8f c) {
  return __builtin_amdgcn_wmma_f32_16x16x32_bf16(false, a, false, b,
                                                 (short)0, c, false, false);
}

__global__ __launch_bounds__(THREADS)
void edge_mlp_wmma(const float* __restrict__ x,  const float* __restrict__ e,
                   const float* __restrict__ nf, const float* __restrict__ ef,
                   const int* __restrict__ src,  const int* __restrict__ dst,
                   const float* __restrict__ W1, const float* __restrict__ b1,
                   const float* __restrict__ W2, const float* __restrict__ b2,
                   const float* __restrict__ W3, const float* __restrict__ b3,
                   float* __restrict__ out, int nE)
{
  extern __shared__ unsigned short sm[];
  unsigned short* sA  = sm + OFF_A;
  unsigned short* sW1 = sm + OFF_W1;
  unsigned short* sH1 = sm + OFF_H1;
  unsigned short* sW2 = sm + OFF_W2;
  unsigned short* sH2 = sm + OFF_H2;

  const int tid  = threadIdx.x;
  const int lane = tid & 31;
  const int wave = tid >> 5;
  const long long blockBase = (long long)blockIdx.x * EPB;

  // ---- Stage W1 as bf16 [128][224], zero-padded K 214..223 (pair granular) ----
  {
    unsigned int* sW1p = (unsigned int*)sW1;
    for (int p = tid; p < HID * (KP / 2); p += THREADS) {
      const int n  = p / (KP / 2);
      const int kp = (p - n * (KP / 2)) * 2;     // even; pair {kp, kp+1}
      unsigned int val = 0u;
      if (kp < CDIM) {                            // CDIM even -> pair fully valid
        const float2 w = *(const float2*)(W1 + n * CDIM + kp);
        val = pack2bf(w.x, w.y);
      }
      sW1p[p] = val;
    }
  }
  // ---- Stage W2 as bf16 [32][128] (pair granular) ----
  {
    unsigned int* sW2p = (unsigned int*)sW2;
    for (int p = tid; p < H2 * HID / 2; p += THREADS) {
      const float2 w = *(const float2*)(W2 + 2 * p);
      sW2p[p] = pack2bf(w.x, w.y);
    }
  }

  // ---- Gather+concat 16 edges per wave into bf16 A tile [128][224] ----
  // Each lane handles one aligned float2 -> one packed dword LDS store.
  for (int i = 0; i < 16; ++i) {
    const int el = wave * 16 + i;
    long long g = blockBase + el;
    if (g >= nE) g = nE - 1;               // grid divides evenly; safety clamp
    const int s = src[g], d = dst[g];
    const float* xs = x + (long long)s * 64;
    const float* xd = x + (long long)d * 64;
    const float* ee = e + g * 64;
    unsigned int* row32 = (unsigned int*)(sA + el * KP);
    // Big segments: lane covers bf16 cols {2*lane, 2*lane+1}
    const float2 v0 = *(const float2*)(xs + 2 * lane);
    row32[lane] = pack2bf(v0.x, v0.y);
    const float2 v1 = *(const float2*)(xd + 2 * lane);
    row32[32 + lane] = pack2bf(v1.x, v1.y);
    const float2 v2 = *(const float2*)(ee + 2 * lane);
    row32[64 + lane] = pack2bf(v2.x, v2.y);
    // Small segments (u32 index c covers bf16 cols {2c, 2c+1}):
    if (lane < 5) {            // nodeftrs[src]: cols 192..201
      const float2 t = *(const float2*)(nf + (long long)s * 10 + 2 * lane);
      row32[96 + lane] = pack2bf(t.x, t.y);
    } else if (lane < 10) {    // nodeftrs[dst]: cols 202..211
      const float2 t = *(const float2*)(nf + (long long)d * 10 + 2 * (lane - 5));
      row32[101 + (lane - 5)] = pack2bf(t.x, t.y);
    } else if (lane == 10) {   // edgeftrs: cols 212..213
      const float2 t = *(const float2*)(ef + g * 2);
      row32[106] = pack2bf(t.x, t.y);
    } else if (lane < 16) {    // zero pad: cols 214..223
      row32[107 + (lane - 11)] = 0u;
    }
  }
  __syncthreads();

  // ---- Layer 1: [128,224] x [224,128]^T -> bias+ReLU -> sH1 bf16 ----
  // Wave tiling: 4 M-groups x 2 N-groups; each wave: 2 M-tiles x 4 N-tiles.
  {
    const int mg = (wave & 3) * 32;   // 2 tiles of 16 rows
    const int ng = (wave >> 2) * 64;  // 4 tiles of 16 cols
    v8f acc[2][4] = {};
#pragma unroll
    for (int ks = 0; ks < KP / 32; ++ks) {
      v16bf a0 = load_a_frag(sA, mg,      ks * 32, KP);
      v16bf a1 = load_a_frag(sA, mg + 16, ks * 32, KP);
#pragma unroll
      for (int nt = 0; nt < 4; ++nt) {
        v16bf b = load_b_frag(sW1, ng + nt * 16, ks * 32, KP);
        acc[0][nt] = wmma_bf16(a0, b, acc[0][nt]);
        acc[1][nt] = wmma_bf16(a1, b, acc[1][nt]);
      }
    }
#pragma unroll
    for (int mi = 0; mi < 2; ++mi) {
      const int mrow = mg + mi * 16 + ((lane < 16) ? 0 : 8);
#pragma unroll
      for (int nt = 0; nt < 4; ++nt) {
        const int n0 = ng + nt * 16 + (lane & 15);
        const float bias = b1[n0];
#pragma unroll
        for (int r = 0; r < 8; ++r) {
          float v = acc[mi][nt][r] + bias;
          v = v > 0.0f ? v : 0.0f;
          sH1[(mrow + r) * HID + n0] = f2bf(v);
        }
      }
    }
  }
  __syncthreads();

  // ---- Layer 2: [128,128] x [128,32]^T -> bias+ReLU -> sH2 bf16 ----
  {
    const int mtile = wave * 16;
    v8f acc[2] = {};
#pragma unroll
    for (int ks = 0; ks < HID / 32; ++ks) {
      v16bf a = load_a_frag(sH1, mtile, ks * 32, HID);
#pragma unroll
      for (int nt = 0; nt < 2; ++nt) {
        v16bf b = load_b_frag(sW2, nt * 16, ks * 32, HID);
        acc[nt] = wmma_bf16(a, b, acc[nt]);
      }
    }
    const int mrow = mtile + ((lane < 16) ? 0 : 8);
#pragma unroll
    for (int nt = 0; nt < 2; ++nt) {
      const int n0 = nt * 16 + (lane & 15);
      const float bias = b2[n0];
#pragma unroll
      for (int r = 0; r < 8; ++r) {
        float v = acc[nt][r] + bias;
        v = v > 0.0f ? v : 0.0f;
        sH2[(mrow + r) * H2 + n0] = f2bf(v);
      }
    }
  }
  __syncthreads();

  // ---- Layer 3: [128,32] x [32,2]^T in VALU (one thread per (edge,class)) ----
  {
    const int el = tid >> 1;
    const int c  = tid & 1;
    float sum = b3[c];
    const float* w3 = W3 + c * H2;
    const unsigned short* h = sH2 + el * H2;
#pragma unroll
    for (int q = 0; q < 4; ++q) {
      const v4u hv = *(const v4u*)(h + q * 8);
#pragma unroll
      for (int j = 0; j < 4; ++j) {
        const unsigned int w = hv[j];
        const float lo = __uint_as_float(w << 16);
        const float hi = __uint_as_float(w & 0xFFFF0000u);
        const int k = q * 8 + j * 2;
        sum += lo * w3[k] + hi * w3[k + 1];
      }
    }
    const long long g = blockBase + el;
    if (g < nE) out[g * 2 + c] = sum;
  }
}

extern "C" void kernel_launch(void* const* d_in, const int* in_sizes, int n_in,
                              void* d_out, int out_size, void* d_ws, size_t ws_size,
                              hipStream_t stream) {
  const float* x  = (const float*)d_in[0];
  const float* e  = (const float*)d_in[1];
  const float* nf = (const float*)d_in[2];
  const float* ef = (const float*)d_in[3];
  const int*   src = (const int*)d_in[4];
  const int*   dst = (const int*)d_in[5];
  const float* W1 = (const float*)d_in[6];
  const float* b1 = (const float*)d_in[7];
  const float* W2 = (const float*)d_in[8];
  const float* b2 = (const float*)d_in[9];
  const float* W3 = (const float*)d_in[10];
  const float* b3 = (const float*)d_in[11];
  float* out = (float*)d_out;

  const int nE = in_sizes[4];                 // number of edges
  const int blocks = (nE + EPB - 1) / EPB;    // 12500 for the reference sizes

  (void)hipFuncSetAttribute((const void*)edge_mlp_wmma,
                            hipFuncAttributeMaxDynamicSharedMemorySize,
                            SMEM_BYTES);

  edge_mlp_wmma<<<blocks, THREADS, SMEM_BYTES, stream>>>(
      x, e, nf, ef, src, dst, W1, b1, W2, b2, W3, b3, out, nE);
}
